// AttentionHyp_41472204210320
// MI455X (gfx1250) — compile-verified, matrix operand order checked by reading, not run on hardware
//
#include <hip/hip_runtime.h>
#include <hip/hip_bf16.h>
#include <math.h>

typedef _Float16 v16h __attribute__((ext_vector_type(16)));
typedef float    v8f  __attribute__((ext_vector_type(8)));

#define Bsz 64
#define Dsz 1024
#define Lsz 1024
#define EPSF 1e-15f
#define PROJ_EPSF 4e-3f

__device__ __forceinline__ float artanh_f(float x) {
    x = fminf(fmaxf(x, -1.0f + 1e-7f), 1.0f - 1e-7f);
    return 0.5f * logf((1.0f + x) / (1.0f - x));
}

__device__ __forceinline__ float wave_sum(float v) {
    #pragma unroll
    for (int off = 16; off > 0; off >>= 1) v += __shfl_xor(v, off, 32);
    return v;
}

// blockDim.x == 256 assumed
__device__ __forceinline__ float block_sum(float v, float* red) {
    int tid = threadIdx.x;
    red[tid] = v; __syncthreads();
    #pragma unroll
    for (int o = 128; o > 0; o >>= 1) {
        if (tid < o) red[tid] += red[tid + o];
        __syncthreads();
    }
    float r = red[0]; __syncthreads();
    return r;
}
__device__ __forceinline__ float block_max(float v, float* red) {
    int tid = threadIdx.x;
    red[tid] = v; __syncthreads();
    #pragma unroll
    for (int o = 128; o > 0; o >>= 1) {
        if (tid < o) red[tid] = fmaxf(red[tid], red[tid + o]);
        __syncthreads();
    }
    float r = red[0]; __syncthreads();
    return r;
}

// Stage 1024 floats (4 KB) global -> LDS with the CDNA5 async-LDS path.
// 256 threads, 16 B per lane.  GVS addressing: mem = SADDR(sgpr64) + VADDR(vgpr32).
__device__ __forceinline__ void async_stage_1k(const float* gsrc, float* sdst) {
    uint32_t lds = (uint32_t)(uintptr_t)sdst + threadIdx.x * 16u;
    uint32_t off = threadIdx.x * 16u;
    asm volatile("global_load_async_to_lds_b128 %0, %1, %2"
                 :: "v"(lds), "v"(off), "s"((uint64_t)(uintptr_t)gsrc)
                 : "memory");
}
__device__ __forceinline__ void async_wait0() {
    asm volatile("s_wait_asynccnt 0" ::: "memory");
}

// Load 16 halves for one lane of a 16x16x32 f16 WMMA operand.
// p points at row_base + k0 + 8*g.  halves[0..7] = p[0..7], halves[8..15] = p[16..23].
__device__ __forceinline__ v16h ld16(const float* p) {
    const float4* q = (const float4*)p;
    float4 x0 = q[0], x1 = q[1], x2 = q[4], x3 = q[5];
    v16h r;
    r[0]=(_Float16)x0.x; r[1]=(_Float16)x0.y; r[2]=(_Float16)x0.z; r[3]=(_Float16)x0.w;
    r[4]=(_Float16)x1.x; r[5]=(_Float16)x1.y; r[6]=(_Float16)x1.z; r[7]=(_Float16)x1.w;
    r[8]=(_Float16)x2.x; r[9]=(_Float16)x2.y; r[10]=(_Float16)x2.z; r[11]=(_Float16)x2.w;
    r[12]=(_Float16)x3.x; r[13]=(_Float16)x3.y; r[14]=(_Float16)x3.z; r[15]=(_Float16)x3.w;
    return r;
}

// -------- Kernel 0: zero the T accumulator --------
__global__ void k_zeroT(float* T) {
    T[blockIdx.x * 256 + threadIdx.x] = 0.0f;
}

// -------- Kernel 1: Q = query(64x1024) * W_in^T  via WMMA --------
__global__ void k_gemm_qin(const float* __restrict__ A, const float* __restrict__ W,
                           float* __restrict__ Qo) {
    int mt = blockIdx.x, nt = blockIdx.y;          // 4 x 64 tiles
    int lane = threadIdx.x & 31;
    int g = lane >> 4, lp = lane & 15;
    const float* arow = A + (size_t)(mt * 16 + lp) * Dsz;
    const float* brow = W + (size_t)(nt * 16 + lp) * Dsz;   // B[k][n] = W[n][k], k contiguous
    v8f acc = {};
    for (int k0 = 0; k0 < Dsz; k0 += 32) {
        v16h a = ld16(arow + k0 + 8 * g);
        v16h b = ld16(brow + k0 + 8 * g);
        acc = __builtin_amdgcn_wmma_f32_16x16x32_f16(false, a, false, b,
                                                     (short)0, acc, false, false);
    }
    int n = nt * 16 + lp;
    #pragma unroll
    for (int r = 0; r < 8; ++r) {
        int m = mt * 16 + r + 8 * g;               // C layout: lanes 0-15 M=r, 16-31 M=8+r
        Qo[(size_t)m * Dsz + n] = acc[r];
    }
}

// -------- Kernel 2: scores + softmax + hyperbolic aw + bt --------
__global__ void k_scores(const float* __restrict__ ctx, const float* __restrict__ Q,
                         const float* __restrict__ dt, const float* __restrict__ cp,
                         const float* __restrict__ ab, float* __restrict__ aw_out,
                         float* __restrict__ bt_out, float* __restrict__ Bn_out) {
    __shared__ float s_q[Dsz];
    __shared__ float s[Lsz];
    __shared__ float red[256];
    int b = blockIdx.x;
    int tid = threadIdx.x, lane = tid & 31, wave = tid >> 5;
    float c = cp[0], sc = sqrtf(c);
    float maxn = (1.0f - PROJ_EPSF) / sc;

    async_stage_1k(Q + (size_t)b * Dsz, s_q);      // stage query row once
    async_wait0();
    __syncthreads();

    const float4* qrow = (const float4*)s_q;
    for (int l = wave; l < Lsz; l += 8) {
        const float4* crow = (const float4*)(ctx + ((size_t)b * Lsz + l) * Dsz);
        float sum = 0.0f;
        #pragma unroll
        for (int j = 0; j < 8; ++j) {
            int idx = lane + 32 * j;
            float4 x = crow[idx], y = qrow[idx];
            sum += x.x * y.x + x.y * y.y + x.z * y.z + x.w * y.w;
        }
        sum = wave_sum(sum);
        if (lane == 0) s[l] = sum;
    }
    __syncthreads();

    // softmax over L (each thread owns 4)
    float mx = -INFINITY;
    #pragma unroll
    for (int i = 0; i < 4; ++i) mx = fmaxf(mx, s[tid * 4 + i]);
    mx = block_max(mx, red);
    float e[4]; float lsum = 0.0f;
    #pragma unroll
    for (int i = 0; i < 4; ++i) { e[i] = expf(s[tid * 4 + i] - mx); lsum += e[i]; }
    float tot = block_sum(lsum, red);
    float inv = 1.0f / tot;
    float n2acc = 0.0f;
    #pragma unroll
    for (int i = 0; i < 4; ++i) { e[i] *= inv; n2acc += e[i] * e[i]; }
    // expmap0 + project (norm over L)
    float n = fmaxf(sqrtf(block_sum(n2acc, red)), EPSF);
    float f = tanhf(sc * n) / (sc * n);
    float nn = f * n;
    if (nn > maxn) f *= maxn / nn;
    #pragma unroll
    for (int i = 0; i < 4; ++i) aw_out[b * Lsz + tid * 4 + i] = e[i] * f;

    // bt = project(expmap0(exp(-ab*dt)))
    float abb = ab[b];
    float bt0[4]; float bacc = 0.0f;
    #pragma unroll
    for (int i = 0; i < 4; ++i) {
        bt0[i] = expf(-abb * dt[b * Lsz + tid * 4 + i]);
        bacc += bt0[i] * bt0[i];
    }
    float bn = fmaxf(sqrtf(block_sum(bacc, red)), EPSF);
    float fb = tanhf(sc * bn) / (sc * bn);
    float nnb = fb * bn;
    if (nnb > maxn) fb *= maxn / nnb;
    #pragma unroll
    for (int i = 0; i < 4; ++i) bt_out[b * Lsz + tid * 4 + i] = bt0[i] * fb;
    if (tid == 0) Bn_out[b] = fmaxf(fb * bn, EPSF);
}

// -------- Kernel 3: per-(b,d) L-reductions + scalar chain --------
__global__ void k_bdstats(const float* __restrict__ ctx, const float* __restrict__ aw,
                          const float* __restrict__ bt, const float* __restrict__ cp,
                          const float* __restrict__ aev, const float* __restrict__ Bn,
                          float* __restrict__ P, float* __restrict__ Qp,
                          float* __restrict__ Qn) {
    __shared__ float s_aw[Lsz];
    __shared__ float s_bt[Lsz];
    int b = blockIdx.x, d = blockIdx.y * 256 + threadIdx.x;

    async_stage_1k(aw + (size_t)b * Lsz, s_aw);
    async_stage_1k(bt + (size_t)b * Lsz, s_bt);
    async_wait0();
    __syncthreads();

    const float* cb = ctx + (size_t)b * Lsz * Dsz + d;
    float S0 = 0, S1 = 0, S2p = 0, S2n = 0, S3p = 0, S3n = 0;
    for (int l = 0; l < Lsz; ++l) {
        if ((l & 7) == 0) __builtin_prefetch(cb + (size_t)(l + 16) * Dsz, 0, 1);
        float x = cb[(size_t)l * Dsz];
        float a = s_aw[l], w = s_bt[l];
        float ax = a * x;
        S0 += x * x;
        S1 += ax * ax;
        float awx = ax * w;
        float s2 = awx * awx;     // (aw*bt*ctx)^2
        float s3 = ax * ax * w;   // aw^2*bt*ctx^2
        if (x > 0.0f) { S2p += s2; S3p += s3; } else { S2n += s2; S3n += s3; }
    }
    float c = cp[0], sc = sqrtf(c);
    float maxn = (1.0f - PROJ_EPSF) / sc;
    float ae = aev[b];

    float xn  = fmaxf(sqrtf(S0), EPSF);
    float wxn = fmaxf(sqrtf(S1), EPSF);
    // mix = mobius_pointwise_mul(aw, ctx) -> g1 * aw * ctx
    float g1 = tanhf(wxn / xn * artanh_f(sc * xn)) / (wxn * sc);
    float n1 = fabsf(g1) * wxn;
    float g2 = (n1 > maxn) ? g1 * (maxn / n1) : g1;                 // project
    // tmp = mobius_pointwise_mul(ae, mix_p)
    float xn2  = fmaxf(fabsf(g2) * wxn, EPSF);
    float wxn2 = fmaxf(fabsf(ae) * fabsf(g2) * wxn, EPSF);
    float f1 = tanhf(wxn2 / xn2 * artanh_f(sc * xn2)) * ae * g2 / (wxn2 * sc);
    float n3 = fabsf(f1) * wxn;
    float f2 = (n3 > maxn) ? f1 * (maxn / n3) : f1;                 // project
    // tmp2 = mobius_pointwise_mul(tmp_p, bt)
    float W3   = fmaxf(sqrtf(S2p + S2n), EPSF);
    float xn3  = Bn[b];
    float wxn3 = fmaxf(fabsf(f2) * W3, EPSF);
    float f3 = tanhf(wxn3 / xn3 * artanh_f(sc * xn3)) * f2 / (wxn3 * sc);
    float n4 = fabsf(f3) * W3;
    float f4 = (n4 > maxn) ? f3 * (maxn / n4) : f3;                 // project (== h4')
    // mobius_add(mix_p, relu(tmp2_p)) -> aw*ctx*(P0 + Q0*bt on region sign(f4*ctx)>0)
    float S2s = (f4 > 0.0f) ? S2p : S2n;
    float S3s = (f4 > 0.0f) ? S3p : S3n;
    float x2 = g2 * g2 * S1;
    float y2 = f4 * f4 * S2s;
    float xy = g2 * f4 * S3s;
    float cA  = 1.0f + 2.0f * c * xy + c * y2;
    float cB  = 1.0f - c * x2;
    float den = fmaxf(1.0f + 2.0f * c * xy + c * c * x2 * y2, EPSF);
    float P0 = cA * g2 / den;
    float Q0 = cB * f4 / den;
    // final project via closed-form norm
    float n5 = fmaxf(sqrtf(fmaxf(P0 * P0 * S1 + 2.0f * P0 * Q0 * S3s + Q0 * Q0 * S2s, 0.0f)), EPSF);
    float s = (n5 > maxn) ? maxn / n5 : 1.0f;
    int idx = b * Dsz + d;
    P[idx]  = s * P0;
    Qp[idx] = (f4 > 0.0f) ? s * Q0 : 0.0f;
    Qn[idx] = (f4 < 0.0f) ? s * Q0 : 0.0f;
}

// -------- Kernel 4: T[b,l] = sum_d mix_final^2 (atomic D-reduction) --------
__global__ void k_T(const float* __restrict__ ctx, const float* __restrict__ aw,
                    const float* __restrict__ bt, const float* __restrict__ P,
                    const float* __restrict__ Qp, const float* __restrict__ Qn,
                    float* __restrict__ T) {
    __shared__ float s_aw[Lsz];
    __shared__ float s_bt[Lsz];
    int b = blockIdx.x, tid = threadIdx.x, d = blockIdx.y * 256 + tid;
    int lane = tid & 31;

    async_stage_1k(aw + (size_t)b * Lsz, s_aw);
    async_stage_1k(bt + (size_t)b * Lsz, s_bt);
    async_wait0();
    __syncthreads();

    int idx = b * Dsz + d;
    float Pv = P[idx], Qpv = Qp[idx], Qnv = Qn[idx];
    const float* cb = ctx + (size_t)b * Lsz * Dsz + d;
    for (int l = 0; l < Lsz; ++l) {
        if ((l & 7) == 0) __builtin_prefetch(cb + (size_t)(l + 16) * Dsz, 0, 1);
        float x = cb[(size_t)l * Dsz];
        float m = s_aw[l] * x * (Pv + ((x > 0.0f) ? Qpv : Qnv) * s_bt[l]);
        float v = wave_sum(m * m);
        if (lane == 0) atomicAdd(&T[b * Lsz + l], v);
    }
}

// -------- Kernel 5: den[b] = clamp(sum_l (lam-1)) --------
__global__ void k_den(const float* __restrict__ T, const float* __restrict__ cp,
                      float* __restrict__ denb) {
    __shared__ float red[256];
    int b = blockIdx.x, tid = threadIdx.x;
    float c = cp[0];
    float acc = 0.0f;
    #pragma unroll
    for (int i = 0; i < 4; ++i) {
        float lam = 2.0f / fmaxf(1.0f - c * T[b * Lsz + tid * 4 + i], EPSF);
        acc += lam - 1.0f;
    }
    float dn = block_sum(acc, red);
    if (tid == 0) {
        dn = (dn >= 0.0f) ? fmaxf(dn, 1e-10f) : fminf(dn, -1e-10f);
        denb[b] = dn;
    }
}

// -------- Kernel 6: nom[b,d] = sum_l lam * mix_final --------
__global__ void k_nom(const float* __restrict__ ctx, const float* __restrict__ aw,
                      const float* __restrict__ bt, const float* __restrict__ T,
                      const float* __restrict__ P, const float* __restrict__ Qp,
                      const float* __restrict__ Qn, const float* __restrict__ cp,
                      float* __restrict__ nom) {
    __shared__ float s_aw[Lsz];
    __shared__ float s_bt[Lsz];
    __shared__ float s_lam[Lsz];
    int b = blockIdx.x, tid = threadIdx.x, d = blockIdx.y * 256 + tid;
    int idx = b * Dsz + d;
    float c = cp[0];

    async_stage_1k(aw + (size_t)b * Lsz, s_aw);
    async_stage_1k(bt + (size_t)b * Lsz, s_bt);
    async_stage_1k(T + (size_t)b * Lsz, s_lam);    // raw T staged, convert below
    async_wait0();
    __syncthreads();
    #pragma unroll
    for (int i = 0; i < 4; ++i) {
        int l = tid * 4 + i;
        s_lam[l] = 2.0f / fmaxf(1.0f - c * s_lam[l], EPSF);
    }
    __syncthreads();

    float Pv = P[idx], Qpv = Qp[idx], Qnv = Qn[idx];
    const float* cb = ctx + (size_t)b * Lsz * Dsz + d;
    float acc = 0.0f;
    for (int l = 0; l < Lsz; ++l) {
        if ((l & 7) == 0) __builtin_prefetch(cb + (size_t)(l + 16) * Dsz, 0, 1);
        float x = cb[(size_t)l * Dsz];
        float m = s_aw[l] * x * (Pv + ((x > 0.0f) ? Qpv : Qnv) * s_bt[l]);
        acc += s_lam[l] * m;
    }
    nom[idx] = acc;
}

// -------- Kernel 7: midpoint (mobius scalar 0.5) + logmap0 --------
__global__ void k_mid(const float* __restrict__ nom, const float* __restrict__ denb,
                      const float* __restrict__ cp, float* __restrict__ mixl) {
    __shared__ float red[256];
    int b = blockIdx.x, tid = threadIdx.x;
    float c = cp[0], sc = sqrtf(c);
    float dn = denb[b];
    float tm[4]; float acc = 0.0f;
    #pragma unroll
    for (int i = 0; i < 4; ++i) {
        tm[i] = nom[b * Dsz + tid * 4 + i] / dn;
        acc += tm[i] * tm[i];
    }
    float n = fmaxf(sqrtf(block_sum(acc, red)), EPSF);
    float r = tanhf(0.5f * artanh_f(sc * n)) / (sc * n);   // mobius_scalar_mul(0.5)
    float n2 = fmaxf(fabsf(r) * n, EPSF);
    float lg = artanh_f(sc * n2) / (sc * n2);              // logmap0
    #pragma unroll
    for (int i = 0; i < 4; ++i)
        mixl[b * Dsz + tid * 4 + i] = lg * r * tm[i];
}

// -------- Kernel 8: out = tanh([mixl, Q] * W_out^T)  via WMMA, K=2048 --------
__global__ void k_gemm_out(const float* __restrict__ mixl, const float* __restrict__ Q,
                           const float* __restrict__ W, float* __restrict__ out) {
    int mt = blockIdx.x, nt = blockIdx.y;          // 4 x 64 tiles
    int lane = threadIdx.x & 31;
    int g = lane >> 4, lp = lane & 15;
    int row = mt * 16 + lp;
    const float* brow = W + (size_t)(nt * 16 + lp) * 2048;   // W_out[n][k], k contiguous
    v8f acc = {};
    for (int k0 = 0; k0 < 2048; k0 += 32) {
        int ka = k0 + 8 * g;                        // 8-blocks never straddle k=1024
        const float* ap = (ka < 1024) ? (mixl + (size_t)row * Dsz + ka)
                                      : (Q + (size_t)row * Dsz + (ka - 1024));
        v16h a = ld16(ap);
        v16h b = ld16(brow + ka);
        acc = __builtin_amdgcn_wmma_f32_16x16x32_f16(false, a, false, b,
                                                     (short)0, acc, false, false);
    }
    int n = nt * 16 + lp;
    #pragma unroll
    for (int r = 0; r < 8; ++r) {
        int m = mt * 16 + r + 8 * g;
        out[(size_t)m * Dsz + n] = tanhf(acc[r]);
    }
}

extern "C" void kernel_launch(void* const* d_in, const int* in_sizes, int n_in,
                              void* d_out, int out_size, void* d_ws, size_t ws_size,
                              hipStream_t stream) {
    const float* query = (const float*)d_in[0];   // (64,1,1024)
    const float* ctx   = (const float*)d_in[1];   // (64,1024,1024)
    const float* dt    = (const float*)d_in[2];   // (64,1,1024)
    const float* cp    = (const float*)d_in[3];   // (1,)
    const float* W_in  = (const float*)d_in[4];   // (1024,1024)
    const float* W_out = (const float*)d_in[5];   // (1024,2048)
    const float* ae    = (const float*)d_in[6];   // (64,1,1)
    const float* ab    = (const float*)d_in[7];   // (64,1,1)

    float* out = (float*)d_out;                   // out (64*1024) then aw (64*1024)
    float* aw  = out + Bsz * Dsz;

    float* w    = (float*)d_ws;
    float* Q    = w;                // 65536
    float* BT   = Q    + 65536;     // 65536
    float* BN   = BT   + 65536;     // 64
    float* PP   = BN   + 64;        // 65536
    float* QP   = PP   + 65536;     // 65536
    float* QN   = QP   + 65536;     // 65536
    float* TT   = QN   + 65536;     // 65536
    float* NOM  = TT   + 65536;     // 65536
    float* MIXL = NOM  + 65536;     // 65536
    float* DENB = MIXL + 65536;     // 64

    k_zeroT   <<<256, 256, 0, stream>>>(TT);
    k_gemm_qin<<<dim3(4, 64), 32, 0, stream>>>(query, W_in, Q);
    k_scores  <<<64, 256, 0, stream>>>(ctx, Q, dt, cp, ab, aw, BT, BN);
    k_bdstats <<<dim3(64, 4), 256, 0, stream>>>(ctx, aw, BT, cp, ae, BN, PP, QP, QN);
    k_T       <<<dim3(64, 4), 256, 0, stream>>>(ctx, aw, BT, PP, QP, QN, TT);
    k_den     <<<64, 256, 0, stream>>>(TT, cp, DENB);
    k_nom     <<<dim3(64, 4), 256, 0, stream>>>(ctx, aw, BT, TT, PP, QP, QN, cp, NOM);
    k_mid     <<<64, 256, 0, stream>>>(NOM, DENB, cp, MIXL);
    k_gemm_out<<<dim3(4, 64), 32, 0, stream>>>(MIXL, Q, W_out, out);
}
